// GPT2SparseDecoderLayer_23167053595228
// MI455X (gfx1250) — matrix operand order repaired and checked by reading, not compiled, not run on hardware
//
#include <hip/hip_runtime.h>
#include <stdint.h>

// ---------------------------------------------------------------------------
// GPT2 sparse decoder layer for MI455X (gfx1250, wave32, WMMA).
// B=2, S=2048, D=1024, H=16, DQ=DK=64, DFF=4096, SPARSE_FACTOR=64.
// All GEMMs run through v_wmma_f32_16x16x32_bf16 (bf16 in, fp32 accum).
// - fragment loads are contiguous 16B LDS reads (ds_load_b128)
// - GEMM pipeline: global_prefetch_b8 pulls tile i+1 into cache while tile
//   i's WMMA chain runs; the post-compute loads then hit in cache and feed
//   LDS stores immediately (no long-lived registers -> no scratch spills).
// - double-buffered LDS: one barrier per K-step.
// ---------------------------------------------------------------------------

typedef __attribute__((ext_vector_type(16))) __bf16 v16bf;
typedef __attribute__((ext_vector_type(8)))  float  v8f;

#define SEQ     2048
#define DIM     1024
#define DFF     4096
#define NTOK    4096      // B * S
#define NHEAD   16
#define HDIM    64
#define BLK     64        // sparse block
#define RES_SCALE 0.70710678118654752f   // 1/sqrt(2)

union Frag16 {            // one WMMA 16-bf16 operand = 2 x 16 bytes
  v16bf v;
  uint4 q[2];
};
union Pack8 {             // 8 bf16 <-> uint4 for transpose staging
  uint4  q;
  __bf16 h[8];
};

// ---------------- WMMA helpers (ISA 7.12.2 layouts) ------------------------

__device__ __forceinline__ v8f wmma_bf16(v16bf a, v16bf b, v8f c) {
  return __builtin_amdgcn_wmma_f32_16x16x32_bf16(
      /*neg_a=*/false, a, /*neg_b=*/false, b,
      /*c_mod=*/(short)0, c, /*reuse_a=*/false, /*reuse_b=*/false);
}

// A fragment: 16x32 bf16 from row-major tile (leading dim ld, 16B-aligned rows)
// lanes 0-15: row=m0+lane,  VGPR0-3 -> K 0..7,  VGPR4-7 -> K 16..23
// lanes16-31: row=m0+lane-16, VGPR0-3 -> K 8..15, VGPR4-7 -> K 24..31
__device__ __forceinline__ v16bf load_a_frag(const __bf16* A, int ld,
                                             int m0, int k0, int lane) {
  const __bf16* p = A + (m0 + (lane & 15)) * ld + k0 + ((lane >> 4) << 3);
  Frag16 f;
  f.q[0] = *(const uint4*)p;          // k = kh .. kh+7
  f.q[1] = *(const uint4*)(p + 16);   // k = 16+kh .. 16+kh+7
  return f.v;
}

// B fragment from a TRANSPOSED tile BT[n][k] (leading dim ld):
// lane = column n0+(lane&15); lanes 0-15 hold K 0..7,16..23, lanes 16-31 K 8..15,24..31.
__device__ __forceinline__ v16bf load_b_fragT(const __bf16* BT, int ld,
                                              int k0, int n0, int lane) {
  const __bf16* p = BT + (n0 + (lane & 15)) * ld + k0 + ((lane >> 4) << 3);
  Frag16 f;
  f.q[0] = *(const uint4*)p;          // k = kh .. kh+7
  f.q[1] = *(const uint4*)(p + 16);   // k = 16+kh .. 16+kh+7
  return f.v;
}

// ---------------- misc pointwise ------------------------------------------

__device__ __forceinline__ float gelu_f(float x) {
  const float c = 0.7978845608028654f;   // sqrt(2/pi)
  float t = tanhf(c * (x + 0.044715f * x * x * x));
  return 0.5f * x * (1.0f + t);
}

__global__ void f32_to_bf16_kernel(const float* __restrict__ in,
                                   __bf16* __restrict__ out, int n) {
  int i = blockIdx.x * 256 + threadIdx.x;
  if (i < n) out[i] = (__bf16)in[i];
}

// LayerNorm over last dim (1024), fp32 in -> bf16 out. 1 block per row.
__global__ void layernorm_kernel(const float* __restrict__ X,
                                 const float* __restrict__ g,
                                 const float* __restrict__ b,
                                 __bf16* __restrict__ out) {
  __shared__ float red[256];
  int row = blockIdx.x;
  const float* x = X + row * DIM;
  float s = 0.f, ss = 0.f;
  for (int i = threadIdx.x; i < DIM; i += 256) {
    float v = x[i]; s += v; ss += v * v;
  }
  red[threadIdx.x] = s; __syncthreads();
  for (int o = 128; o > 0; o >>= 1) {
    if (threadIdx.x < o) red[threadIdx.x] += red[threadIdx.x + o];
    __syncthreads();
  }
  float mu = red[0] * (1.0f / DIM);
  __syncthreads();
  red[threadIdx.x] = ss; __syncthreads();
  for (int o = 128; o > 0; o >>= 1) {
    if (threadIdx.x < o) red[threadIdx.x] += red[threadIdx.x + o];
    __syncthreads();
  }
  float var = red[0] * (1.0f / DIM) - mu * mu;
  float rs = rsqrtf(var + 1e-5f);
  for (int i = threadIdx.x; i < DIM; i += 256)
    out[row * DIM + i] = (__bf16)((x[i] - mu) * rs * g[i] + b[i]);
}

// ---------------- generic tiled WMMA GEMM (prefetch pipeline) --------------
// C[M,N] = act(A[M,K] @ B[K,N] + bias)   (A,B bf16 row-major in global)
// mode 0: out_bf = result                (QKV projections)
// mode 1: out_bf = gelu(result)          (FFN up-proj)
// mode 2: out_f  = resid + scale*result  (attn out-proj, FFN down-proj)
// 256 threads = 8 waves; tile 128x64; K-step 64; one barrier per K-step.

#define GM_TM 128
#define GM_TN 64
#define GM_TK 64
#define A_LD  (GM_TK + 8)   // 72 bf16 -> 144B row stride (16B aligned)
#define B_LD  (GM_TK + 8)

__global__ __launch_bounds__(256, 1)
void gemm_wmma_kernel(const __bf16* __restrict__ A,
                      const __bf16* __restrict__ B,
                      const float* __restrict__ bias,
                      const float* __restrict__ resid,
                      float scale,
                      __bf16* __restrict__ out_bf,
                      float* __restrict__ out_f,
                      int M, int N, int K, int mode) {
  __shared__ __align__(16) __bf16 As [2][GM_TM][A_LD];   // 2 x 18 KB
  __shared__ __align__(16) __bf16 BsT[2][GM_TN][B_LD];   // 2 x  9 KB
  const int tid  = threadIdx.x;
  const int lane = tid & 31;
  const int wave = tid >> 5;
  const int m0 = blockIdx.x * GM_TM;
  const int n0 = blockIdx.y * GM_TN;

  v8f acc[4] = {v8f{}, v8f{}, v8f{}, v8f{}};

  // ring the lines for K-tile k0 into WGP$/L2 (no data return, no counters)
  auto prefetch_tile = [&](int k0) {
#pragma unroll
    for (int it = 0; it < 4; ++it) {
      int idx = tid + it * 256;
      int r   = idx >> 3;
      int c8  = (idx & 7) << 3;
      __builtin_prefetch(&A[(size_t)(m0 + r) * K + k0 + c8], 0, 3);
    }
#pragma unroll
    for (int it = 0; it < 2; ++it) {
      int idx = tid + it * 256;
      int r   = idx >> 3;
      int c8  = (idx & 7) << 3;
      __builtin_prefetch(&B[(size_t)(k0 + r) * N + n0 + c8], 0, 3);
    }
  };
  // copy K-tile k0 global -> LDS buffer buf (loads are short-lived: no spill)
  auto stage_tile = [&](int k0, int buf) {
#pragma unroll
    for (int it = 0; it < 4; ++it) {
      int idx = tid + it * 256;
      int r   = idx >> 3;
      int c8  = (idx & 7) << 3;
      *(uint4*)&As[buf][r][c8] =
          *(const uint4*)&A[(size_t)(m0 + r) * K + k0 + c8];
    }
#pragma unroll
    for (int it = 0; it < 2; ++it) {
      int idx = tid + it * 256;
      int r   = idx >> 3;            // k row of B
      int c8  = (idx & 7) << 3;      // n col group
      Pack8 pk;
      pk.q = *(const uint4*)&B[(size_t)(k0 + r) * N + n0 + c8];
#pragma unroll
      for (int j = 0; j < 8; ++j) BsT[buf][c8 + j][r] = pk.h[j];
    }
  };

  stage_tile(0, 0);
  __syncthreads();

  int buf = 0;
  for (int k0 = 0; k0 < K; k0 += GM_TK) {
    const bool more = (k0 + GM_TK) < K;
    if (more) prefetch_tile(k0 + GM_TK);   // lines fly during WMMA chain

#pragma unroll
    for (int kk = 0; kk < GM_TK; kk += 32) {
      v16bf af = load_a_frag(&As[buf][0][0], A_LD, wave * 16, kk, lane);
      v16bf bfr[4];
#pragma unroll
      for (int t = 0; t < 4; ++t)
        bfr[t] = load_b_fragT(&BsT[buf][0][0], B_LD, kk, t * 16, lane);
#pragma unroll
      for (int t = 0; t < 4; ++t)
        acc[t] = wmma_bf16(af, bfr[t], acc[t]);
    }

    if (more) {
      stage_tile(k0 + GM_TK, buf ^ 1);  // cache hits; writes buf^1 (no reader
      __syncthreads();                  // past previous barrier); publish
      buf ^= 1;
    }
  }

  // epilogue: D layout -> lane = col, VGPR r = row r (+8 for upper lanes)
  const int coll = lane & 15;
  const int rb   = (lane >> 4) * 8;
#pragma unroll
  for (int t = 0; t < 4; ++t) {
    int n = n0 + t * 16 + coll;
    float bv = bias[n];
#pragma unroll
    for (int r = 0; r < 8; ++r) {
      int m = m0 + wave * 16 + rb + r;
      float v = acc[t][r] + bv;
      size_t idx = (size_t)m * N + n;
      if (mode == 2) {
        out_f[idx] = resid[idx] + scale * v;
      } else {
        if (mode == 1) v = gelu_f(v);
        out_bf[idx] = (__bf16)v;
      }
    }
  }
}

// ---------------- block-sparse causal attention ---------------------------
// One workgroup (128 threads / 4 waves) per (batch, head, 64-block).
// Q,K,V are [NTOK, DIM] bf16, head h occupies cols h*64..h*64+63.
// K tile row-major IS the transposed-B layout for Q*K^T; V is staged
// transposed for P*V. All fragment reads are 2x ds_load_b128.

#define T_LD (HDIM + 8)     // 72 bf16 rows, 144B stride

__global__ __launch_bounds__(128, 1)
void attn_block_kernel(const __bf16* __restrict__ Q,
                       const __bf16* __restrict__ K,
                       const __bf16* __restrict__ V,
                       __bf16* __restrict__ CTX) {
  __shared__ __align__(16) __bf16 Qs [BLK][T_LD];   // 9 KB row-major
  __shared__ __align__(16) __bf16 Ks [BLK][T_LD];   // 9 KB row-major (=B^T)
  __shared__ __align__(16) __bf16 VsT[HDIM][T_LD];  // 9 KB transposed
  __shared__ __align__(16) float  Ss [BLK][BLK + 4];// 17 KB scores
  __shared__ __align__(16) __bf16 Ps [BLK][T_LD];   // 9 KB probs

  const int gb    = blockIdx.x;          // B*H*(S/BLK) = 1024
  const int blk   = gb & 31;             // S/BLK = 32
  const int head  = (gb >> 5) & 15;
  const int batch = gb >> 9;
  const int row0 = batch * SEQ + blk * BLK;   // token row in [NTOK, DIM]
  const int col0 = head * HDIM;

  const int tid  = threadIdx.x;          // 128
  const int lane = tid & 31;
  const int wave = tid >> 5;

  // stage Q/K row-major, V transposed: 512 uint4 each over 128 threads
#pragma unroll
  for (int it = 0; it < 4; ++it) {
    int idx = tid + it * 128;
    int r   = idx >> 3;                  // token row in block
    int c8  = (idx & 7) << 3;            // head-dim col group
    size_t gidx = (size_t)(row0 + r) * DIM + col0 + c8;
    *(uint4*)&Qs[r][c8] = *(const uint4*)&Q[gidx];
    *(uint4*)&Ks[r][c8] = *(const uint4*)&K[gidx];
    Pack8 pv;
    pv.q = *(const uint4*)&V[gidx];
#pragma unroll
    for (int j = 0; j < 8; ++j) VsT[c8 + j][r] = pv.h[j];
  }
  __syncthreads();

  // S = (Q @ K^T) / sqrt(64); wave w owns rows [w*16, w*16+16)
  const int mrow = wave * 16;
  {
    v8f acc[4] = {v8f{}, v8f{}, v8f{}, v8f{}};
#pragma unroll
    for (int kk = 0; kk < HDIM; kk += 32) {
      v16bf a = load_a_frag(&Qs[0][0], T_LD, mrow, kk, lane);
      v16bf bfr[4];
#pragma unroll
      for (int t = 0; t < 4; ++t)
        bfr[t] = load_b_fragT(&Ks[0][0], T_LD, kk, t * 16, lane); // B^T = Ks
#pragma unroll
      for (int t = 0; t < 4; ++t)
        acc[t] = wmma_bf16(a, bfr[t], acc[t]);
    }
    int rb = (lane >> 4) * 8;
#pragma unroll
    for (int t = 0; t < 4; ++t) {
      int n = t * 16 + (lane & 15);
#pragma unroll
      for (int r = 0; r < 8; ++r)
        Ss[mrow + rb + r][n] = acc[t][r] * 0.125f;   // 1/sqrt(DQ)
    }
  }
  __syncthreads();

  // causal softmax within the block: row r attends cols 0..r
  if (tid < BLK) {
    int r = tid;
    float mx = -1e30f;
    for (int c = 0; c <= r; ++c) mx = fmaxf(mx, Ss[r][c]);
    float sum = 0.f;
    for (int c = 0; c <= r; ++c) {
      float e = __expf(Ss[r][c] - mx);
      Ss[r][c] = e;
      sum += e;
    }
    float inv = 1.0f / sum;
    for (int c = 0; c < BLK; ++c)
      Ps[r][c] = (__bf16)((c <= r) ? Ss[r][c] * inv : 0.0f);
  }
  __syncthreads();

  // CTX = P @ V   (B^T = VsT)
  {
    v8f acc[4] = {v8f{}, v8f{}, v8f{}, v8f{}};
#pragma unroll
    for (int kk = 0; kk < BLK; kk += 32) {
      v16bf a = load_a_frag(&Ps[0][0], T_LD, mrow, kk, lane);
      v16bf bfr[4];
#pragma unroll
      for (int t = 0; t < 4; ++t)
        bfr[t] = load_b_fragT(&VsT[0][0], T_LD, kk, t * 16, lane);
#pragma unroll
      for (int t = 0; t < 4; ++t)
        acc[t] = wmma_bf16(a, bfr[t], acc[t]);
    }
    int rb = (lane >> 4) * 8;
#pragma unroll
    for (int t = 0; t < 4; ++t) {
      int n = t * 16 + (lane & 15);
#pragma unroll
      for (int r = 0; r < 8; ++r)
        CTX[(size_t)(row0 + mrow + rb + r) * DIM + col0 + n] = (__bf16)acc[t][r];
    }
  }
}

// ---------------------------------------------------------------------------

extern "C" void kernel_launch(void* const* d_in, const int* in_sizes, int n_in,
                              void* d_out, int out_size, void* d_ws, size_t ws_size,
                              hipStream_t stream) {
  (void)in_sizes; (void)n_in; (void)out_size; (void)ws_size;

  const float* x     = (const float*)d_in[0];
  // d_in[1] = mask_sa (bool) -- causality & block mask computed analytically
  const float* Wq  = (const float*)d_in[2];  const float* bq = (const float*)d_in[3];
  const float* Wk  = (const float*)d_in[4];  const float* bk = (const float*)d_in[5];
  const float* Wv  = (const float*)d_in[6];  const float* bv = (const float*)d_in[7];
  const float* Wo  = (const float*)d_in[8];  const float* bo = (const float*)d_in[9];
  const float* ln1g = (const float*)d_in[10]; const float* ln1b = (const float*)d_in[11];
  const float* ln2g = (const float*)d_in[12]; const float* ln2b = (const float*)d_in[13];
  const float* W1  = (const float*)d_in[14]; const float* b1 = (const float*)d_in[15];
  const float* W2  = (const float*)d_in[16]; const float* b2 = (const float*)d_in[17];
  float* out = (float*)d_out;

  // ---- workspace carve-up (bytes) ----
  char* p = (char*)d_ws;
  __bf16* hb   = (__bf16*)(p);                         // 4096x1024 bf16
  __bf16* h2b  = (__bf16*)(p + 8388608);
  __bf16* qb   = (__bf16*)(p + 16777216);
  __bf16* kb   = (__bf16*)(p + 25165824);
  __bf16* vb   = (__bf16*)(p + 33554432);
  __bf16* ctxb = (__bf16*)(p + 41943040);
  __bf16* Wqb  = (__bf16*)(p + 50331648);              // 1024x1024 bf16
  __bf16* Wkb  = (__bf16*)(p + 52428800);
  __bf16* Wvb  = (__bf16*)(p + 54525952);
  __bf16* Wob  = (__bf16*)(p + 56623104);
  __bf16* W1b  = (__bf16*)(p + 58720256);              // 1024x4096 bf16
  __bf16* W2b  = (__bf16*)(p + 67108864);              // 4096x1024 bf16
  __bf16* ff1b = (__bf16*)(p + 75497472);              // 4096x4096 bf16
  float*  x1   = (float*) (p + 109051904);             // 4096x1024 f32

  // ---- weight conversions fp32 -> bf16 ----
  const int WSMALL = DIM * DIM;       // 1,048,576
  const int WBIG   = DIM * DFF;       // 4,194,304
  f32_to_bf16_kernel<<<WSMALL / 256, 256, 0, stream>>>(Wq, Wqb, WSMALL);
  f32_to_bf16_kernel<<<WSMALL / 256, 256, 0, stream>>>(Wk, Wkb, WSMALL);
  f32_to_bf16_kernel<<<WSMALL / 256, 256, 0, stream>>>(Wv, Wvb, WSMALL);
  f32_to_bf16_kernel<<<WSMALL / 256, 256, 0, stream>>>(Wo, Wob, WSMALL);
  f32_to_bf16_kernel<<<WBIG / 256, 256, 0, stream>>>(W1, W1b, WBIG);
  f32_to_bf16_kernel<<<WBIG / 256, 256, 0, stream>>>(W2, W2b, WBIG);

  // ---- sublayer 1: h = LN1(x) ----
  layernorm_kernel<<<NTOK, 256, 0, stream>>>(x, ln1g, ln1b, hb);

  // ---- Q/K/V projections: [4096,1024] @ [1024,1024] ----
  dim3 gqkv(NTOK / GM_TM, DIM / GM_TN);
  gemm_wmma_kernel<<<gqkv, 256, 0, stream>>>(hb, Wqb, bq, nullptr, 0.f, qb, nullptr,
                                             NTOK, DIM, DIM, 0);
  gemm_wmma_kernel<<<gqkv, 256, 0, stream>>>(hb, Wkb, bk, nullptr, 0.f, kb, nullptr,
                                             NTOK, DIM, DIM, 0);
  gemm_wmma_kernel<<<gqkv, 256, 0, stream>>>(hb, Wvb, bv, nullptr, 0.f, vb, nullptr,
                                             NTOK, DIM, DIM, 0);

  // ---- block-diagonal causal attention: 2*16*32 = 1024 blocks ----
  attn_block_kernel<<<2 * NHEAD * (SEQ / BLK), 128, 0, stream>>>(qb, kb, vb, ctxb);

  // ---- out-proj + scaled residual: x1 = x + ctx@Wo*res_scale ----
  gemm_wmma_kernel<<<gqkv, 256, 0, stream>>>(ctxb, Wob, bo, x, RES_SCALE,
                                             nullptr, x1, NTOK, DIM, DIM, 2);

  // ---- sublayer 2: h2 = LN2(x1) ----
  layernorm_kernel<<<NTOK, 256, 0, stream>>>(x1, ln2g, ln2b, h2b);

  // ---- FFN up + GELU: [4096,1024] @ [1024,4096] ----
  dim3 gff1(NTOK / GM_TM, DFF / GM_TN);
  gemm_wmma_kernel<<<gff1, 256, 0, stream>>>(h2b, W1b, b1, nullptr, 0.f,
                                             ff1b, nullptr, NTOK, DFF, DIM, 1);

  // ---- FFN down + scaled residual into d_out ----
  gemm_wmma_kernel<<<gqkv, 256, 0, stream>>>(ff1b, W2b, b2, x1, RES_SCALE,
                                             nullptr, out, NTOK, DIM, DFF, 2);
}
